// PIGNN_29669634081208
// MI455X (gfx1250) — matrix operand order repaired and checked by reading, not compiled, not run on hardware
//
#include <hip/hip_runtime.h>

// ---------------------------------------------------------------------------
// CDNA5 (gfx1250) WMMA GNN: 6-layer message passing, HID=128.
// All GEMMs run on v_wmma_f32_16x16x32_bf16 (f32 accum).
// Weights are pre-packed (fp32 -> bf16) into the native B-fragment layout so
// each wave fetches one 1KB contiguous fragment (32B per lane) per k-step.
// ---------------------------------------------------------------------------

typedef __attribute__((ext_vector_type(16))) __bf16 v16bf;
typedef __attribute__((ext_vector_type(8)))  float  v8f;

union ABfrag { v16bf v; unsigned u[8]; };

// native f32 -> bf16 (RNE) — lets clang use v_cvt_pk_bf16_f32-class ops
__device__ __forceinline__ __bf16 f2bf(float f) { return (__bf16)f; }

__device__ __forceinline__ unsigned pack2bf(float a, float b) {
  unsigned short ua = __builtin_bit_cast(unsigned short, (__bf16)a);
  unsigned short ub = __builtin_bit_cast(unsigned short, (__bf16)b);
  return (unsigned)ua | ((unsigned)ub << 16);
}

__device__ __forceinline__ v8f vzero8() {
  v8f z = {0.f, 0.f, 0.f, 0.f, 0.f, 0.f, 0.f, 0.f};
  return z;
}

// ---------------------------------------------------------------------------
// Wave-level GEMM: D[16*ROWT x 16] += A(LDS bf16 row-major [16*ROWT][Kpad]) * Wp
// Wp is pre-packed B fragments: element (nt,kc,lane,i) at
//   Wp[((nt*KC+kc)*32 + lane)*16 + i]  with  K = kc*32 + (lane<16?0:8)+(i&7)+(i&8?16:0),
//                                            N = nt*16 + (lane&15).
// A fragment per ISA 16-bit A 16x32 layout: lane<16 -> Kbase 0, else Kbase 8;
// VGPR v<4 holds K pair (Kbase+2v), VGPR v>=4 holds K pair (Kbase+16+2(v-4)).
// ---------------------------------------------------------------------------
template <int KC, int ROWT>
__device__ __forceinline__ void gemm_tiles(const __bf16* ldsA, int Kpad,
                                           const __bf16* __restrict__ Wp, int nt,
                                           v8f acc[ROWT]) {
  const int lane = threadIdx.x & 31;
  const int arow = lane & 15;
  const int kb   = (lane < 16) ? 0 : 8;
#pragma unroll
  for (int kc = 0; kc < KC; ++kc) {
    ABfrag b;
    b.v = *(const v16bf*)(Wp + (size_t)(((nt * KC + kc) * 32 + lane) * 16));
    const int base = (kc * 32 + kb) >> 1;   // dword index within row
#pragma unroll
    for (int rt = 0; rt < ROWT; ++rt) {
      const unsigned* ar = (const unsigned*)(ldsA + (size_t)(rt * 16 + arow) * Kpad);
      ABfrag a;
#pragma unroll
      for (int v = 0; v < 4; ++v) {
        a.u[v]     = ar[base + v];
        a.u[v + 4] = ar[base + 8 + v];
      }
      acc[rt] = __builtin_amdgcn_wmma_f32_16x16x32_bf16(
          false, a.v, false, b.v, (short)0, acc[rt], false, false);
    }
  }
}

// bias (+ optional ReLU), write tile to LDS as bf16 [16*ROWT][Hpad]
template <int ROWT>
__device__ __forceinline__ void acc_to_lds(const v8f acc[ROWT], __bf16* ldsH, int Hpad,
                                           int nt, const float* __restrict__ bias,
                                           bool relu) {
  const int lane = threadIdx.x & 31;
  const int col  = nt * 16 + (lane & 15);
  const int rb   = (lane < 16) ? 0 : 8;
  const float bv = bias[col];
#pragma unroll
  for (int rt = 0; rt < ROWT; ++rt) {
#pragma unroll
    for (int r = 0; r < 8; ++r) {
      float v = acc[rt][r] + bv;
      if (relu) v = fmaxf(v, 0.f);
      ldsH[(size_t)(rt * 16 + rb + r) * Hpad + col] = f2bf(v);
    }
  }
}

// ---------------------------------------------------------------------------
// Weight packer: fp32 [Kin][Nout] row-major -> bf16 B fragments, zero padded.
// ---------------------------------------------------------------------------
__global__ __launch_bounds__(256) void pack_w(const float* __restrict__ W,
                                              __bf16* __restrict__ out,
                                              int Kin, int Nout, int KC, int NT) {
  int tid = blockIdx.x * blockDim.x + threadIdx.x;
  int total = NT * KC * 512;
  if (tid >= total) return;
  int i = tid & 15, l = (tid >> 4) & 31, g = tid >> 9;
  int kc = g % KC, nt = g / KC;
  int k = kc * 32 + ((l < 16) ? 0 : 8) + (i & 7) + ((i & 8) ? 16 : 0);
  int n = nt * 16 + (l & 15);
  float v = (k < Kin && n < Nout) ? W[(size_t)k * Nout + n] : 0.f;
  out[tid] = f2bf(v);
}

__global__ __launch_bounds__(256) void zero_f32v4(float4* __restrict__ p, int n4) {
  int i = blockIdx.x * blockDim.x + threadIdx.x;
  if (i < n4) p[i] = make_float4(0.f, 0.f, 0.f, 0.f);
}

// stage a float4 into LDS (bf16), LDS dword index must be even
__device__ __forceinline__ void st4bf(__bf16* lds, int elemIdx, float4 v) {
  uint2 p;
  p.x = pack2bf(v.x, v.y);
  p.y = pack2bf(v.z, v.w);
  *(uint2*)((unsigned*)lds + (elemIdx >> 1)) = p;
}

// ---------------------------------------------------------------------------
// Node encoder: feat = [coords | x[3:9] | 0pad] (K=32), MLP 32->128 ReLU ->128
// 64 nodes / block (4 row tiles), 8 waves = 8 col tiles.
// ---------------------------------------------------------------------------
__global__ __launch_bounds__(256) void enc_nodes(
    const float* __restrict__ x, const float* __restrict__ coords,
    const __bf16* __restrict__ W0p, const float* __restrict__ b0,
    const __bf16* __restrict__ W1p, const float* __restrict__ b1,
    float* __restrict__ h, int N) {
  __shared__ __bf16 ldsA[64 * 32];
  __shared__ __bf16 ldsH[64 * 128];
  const int tid = threadIdx.x;
  const int base = blockIdx.x * 64;
  for (int t = tid; t < 64 * 32; t += 256) {
    int row = t >> 5, j = t & 31, node = base + row;
    float v = 0.f;
    if (node < N && j < 9) v = (j < 3) ? coords[(size_t)node * 3 + j] : x[(size_t)node * 9 + j];
    ldsA[t] = f2bf(v);
  }
  __syncthreads();
  const int w = tid >> 5;
  v8f acc[4];
#pragma unroll
  for (int rt = 0; rt < 4; ++rt) acc[rt] = vzero8();
  gemm_tiles<1, 4>(ldsA, 32, W0p, w, acc);
  acc_to_lds<4>(acc, ldsH, 128, w, b0, true);
  __syncthreads();
  v8f acc2[4];
#pragma unroll
  for (int rt = 0; rt < 4; ++rt) acc2[rt] = vzero8();
  gemm_tiles<4, 4>(ldsH, 128, W1p, w, acc2);
  const int lane = tid & 31;
  const int col = w * 16 + (lane & 15);
  const int rb = (lane < 16) ? 0 : 8;
  const float bv = b1[col];
#pragma unroll
  for (int rt = 0; rt < 4; ++rt)
#pragma unroll
    for (int r = 0; r < 8; ++r) {
      int node = base + rt * 16 + rb + r;
      if (node < N) h[(size_t)node * 128 + col] = acc2[rt][r] + bv;
    }
}

// Edge encoder: edge_attr (K=10 padded to 32) -> e [E,128]
__global__ __launch_bounds__(256) void enc_edges(
    const float* __restrict__ ea,
    const __bf16* __restrict__ W0p, const float* __restrict__ b0,
    const __bf16* __restrict__ W1p, const float* __restrict__ b1,
    float* __restrict__ e, int E) {
  __shared__ __bf16 ldsA[64 * 32];
  __shared__ __bf16 ldsH[64 * 128];
  const int tid = threadIdx.x;
  const int base = blockIdx.x * 64;
  for (int t = tid; t < 64 * 32; t += 256) {
    int row = t >> 5, j = t & 31, ed = base + row;
    float v = 0.f;
    if (ed < E && j < 10) v = ea[(size_t)ed * 10 + j];
    ldsA[t] = f2bf(v);
  }
  __syncthreads();
  const int w = tid >> 5;
  v8f acc[4];
#pragma unroll
  for (int rt = 0; rt < 4; ++rt) acc[rt] = vzero8();
  gemm_tiles<1, 4>(ldsA, 32, W0p, w, acc);
  acc_to_lds<4>(acc, ldsH, 128, w, b0, true);
  __syncthreads();
  v8f acc2[4];
#pragma unroll
  for (int rt = 0; rt < 4; ++rt) acc2[rt] = vzero8();
  gemm_tiles<4, 4>(ldsH, 128, W1p, w, acc2);
  const int lane = tid & 31;
  const int col = w * 16 + (lane & 15);
  const int rb = (lane < 16) ? 0 : 8;
  const float bv = b1[col];
#pragma unroll
  for (int rt = 0; rt < 4; ++rt)
#pragma unroll
    for (int r = 0; r < 8; ++r) {
      int ed = base + rt * 16 + rb + r;
      if (ed < E) e[(size_t)ed * 128 + col] = acc2[rt][r] + bv;
    }
}

// ---------------------------------------------------------------------------
// Edge update + scatter-add: m = MLP([h[src]|h[dst]|e], 384->128 ReLU ->128),
// agg[dst] += m  (native f32 atomics). 32 edges / block (2 row tiles).
// ---------------------------------------------------------------------------
__global__ __launch_bounds__(256) void edge_mp(
    const float* __restrict__ h, const float* __restrict__ e,
    const int* __restrict__ src, const int* __restrict__ dst,
    const __bf16* __restrict__ W0p, const float* __restrict__ b0,
    const __bf16* __restrict__ W1p, const float* __restrict__ b1,
    float* __restrict__ agg, int E) {
  __shared__ int sSrc[32], sDst[32];
  __shared__ __bf16 ldsA[32 * 384];
  __shared__ __bf16 ldsH[32 * 128];
  const int tid = threadIdx.x;
  const int base = blockIdx.x * 32;
  if (tid < 32) {
    int ed = base + tid;
    sSrc[tid] = (ed < E) ? src[ed] : -1;
    sDst[tid] = (ed < E) ? dst[ed] : -1;
  }
  __syncthreads();
  // float4 gather: 96 float4 per row
  for (int t = tid; t < 32 * 96; t += 256) {
    int row = t / 96;
    int c = (t - row * 96) * 4;
    float4 v = make_float4(0.f, 0.f, 0.f, 0.f);
    int s = sSrc[row];
    if (s >= 0) {
      if (c < 128)       v = *(const float4*)(h + (size_t)s * 128 + c);
      else if (c < 256)  v = *(const float4*)(h + (size_t)sDst[row] * 128 + (c - 128));
      else               v = *(const float4*)(e + (size_t)(base + row) * 128 + (c - 256));
    }
    st4bf(ldsA, row * 384 + c, v);
  }
  __syncthreads();
  const int w = tid >> 5;
  v8f acc[2];
  acc[0] = vzero8(); acc[1] = vzero8();
  gemm_tiles<12, 2>(ldsA, 384, W0p, w, acc);
  acc_to_lds<2>(acc, ldsH, 128, w, b0, true);
  __syncthreads();
  v8f acc2[2];
  acc2[0] = vzero8(); acc2[1] = vzero8();
  gemm_tiles<4, 2>(ldsH, 128, W1p, w, acc2);
  const int lane = tid & 31;
  const int col = w * 16 + (lane & 15);
  const int rb = (lane < 16) ? 0 : 8;
  const float bv = b1[col];
#pragma unroll
  for (int rt = 0; rt < 2; ++rt)
#pragma unroll
    for (int r = 0; r < 8; ++r) {
      int d = sDst[rt * 16 + rb + r];
      if (d >= 0)
        __hip_atomic_fetch_add(&agg[(size_t)d * 128 + col], acc2[rt][r] + bv,
                               __ATOMIC_RELAXED, __HIP_MEMORY_SCOPE_AGENT);
    }
}

// ---------------------------------------------------------------------------
// Node update: h += MLP([h|agg], 256->128 ReLU ->128). 64 nodes / block.
// ---------------------------------------------------------------------------
__global__ __launch_bounds__(256) void node_mp(
    float* __restrict__ h, const float* __restrict__ agg,
    const __bf16* __restrict__ W0p, const float* __restrict__ b0,
    const __bf16* __restrict__ W1p, const float* __restrict__ b1, int N) {
  __shared__ __bf16 ldsA[64 * 256];
  __shared__ __bf16 ldsH[64 * 128];
  const int tid = threadIdx.x;
  const int base = blockIdx.x * 64;
  // float4 staging: 64 float4 per row ([h | agg])
  for (int t = tid; t < 64 * 64; t += 256) {
    int row = t >> 6, c = (t & 63) * 4, node = base + row;
    float4 v = make_float4(0.f, 0.f, 0.f, 0.f);
    if (node < N)
      v = (c < 128) ? *(const float4*)(h + (size_t)node * 128 + c)
                    : *(const float4*)(agg + (size_t)node * 128 + (c - 128));
    st4bf(ldsA, row * 256 + c, v);
  }
  __syncthreads();
  const int w = tid >> 5;
  v8f acc[4];
#pragma unroll
  for (int rt = 0; rt < 4; ++rt) acc[rt] = vzero8();
  gemm_tiles<8, 4>(ldsA, 256, W0p, w, acc);
  acc_to_lds<4>(acc, ldsH, 128, w, b0, true);
  __syncthreads();
  v8f acc2[4];
#pragma unroll
  for (int rt = 0; rt < 4; ++rt) acc2[rt] = vzero8();
  gemm_tiles<4, 4>(ldsH, 128, W1p, w, acc2);
  const int lane = tid & 31;
  const int col = w * 16 + (lane & 15);
  const int rb = (lane < 16) ? 0 : 8;
  const float bv = b1[col];
#pragma unroll
  for (int rt = 0; rt < 4; ++rt)
#pragma unroll
    for (int r = 0; r < 8; ++r) {
      int node = base + rt * 16 + rb + r;
      if (node < N) {
        size_t idx = (size_t)node * 128 + col;
        h[idx] = h[idx] + acc2[rt][r] + bv;   // residual
      }
    }
}

// ---------------------------------------------------------------------------
// Decoder: 128->128 ReLU -> 64 ReLU -> 15, fused BC masks.
// ---------------------------------------------------------------------------
__global__ __launch_bounds__(256) void decode(
    const float* __restrict__ h,
    const __bf16* __restrict__ W0p, const float* __restrict__ b0,
    const __bf16* __restrict__ W1p, const float* __restrict__ b1,
    const __bf16* __restrict__ W2p, const float* __restrict__ b2,
    const int* __restrict__ bc_disp, const int* __restrict__ bc_rot,
    const int* __restrict__ face_mask,
    float* __restrict__ pred, int N) {
  __shared__ __bf16 ldsA[64 * 128];
  __shared__ __bf16 ldsH[64 * 128];
  __shared__ __bf16 ldsH2[64 * 64];
  const int tid = threadIdx.x;
  const int base = blockIdx.x * 64;
  for (int t = tid; t < 64 * 32; t += 256) {
    int row = t >> 5, c = (t & 31) * 4, node = base + row;
    float4 v = make_float4(0.f, 0.f, 0.f, 0.f);
    if (node < N) v = *(const float4*)(h + (size_t)node * 128 + c);
    st4bf(ldsA, row * 128 + c, v);
  }
  __syncthreads();
  const int w = tid >> 5;
  {
    v8f acc[4];
#pragma unroll
    for (int rt = 0; rt < 4; ++rt) acc[rt] = vzero8();
    gemm_tiles<4, 4>(ldsA, 128, W0p, w, acc);
    acc_to_lds<4>(acc, ldsH, 128, w, b0, true);
  }
  __syncthreads();
  if (w < 4) {
    v8f acc[4];
#pragma unroll
    for (int rt = 0; rt < 4; ++rt) acc[rt] = vzero8();
    gemm_tiles<4, 4>(ldsH, 128, W1p, w, acc);
    acc_to_lds<4>(acc, ldsH2, 64, w, b1, true);
  }
  __syncthreads();
  if (w == 0) {
    v8f acc[4];
#pragma unroll
    for (int rt = 0; rt < 4; ++rt) acc[rt] = vzero8();
    gemm_tiles<2, 4>(ldsH2, 64, W2p, 0, acc);
    const int lane = tid & 31;
    const int col = lane & 15;
    const int rb = (lane < 16) ? 0 : 8;
    if (col < 15) {
      const float bv = b2[col];
#pragma unroll
      for (int rt = 0; rt < 4; ++rt)
#pragma unroll
        for (int r = 0; r < 8; ++r) {
          int node = base + rt * 16 + rb + r;
          if (node < N) {
            float m;
            if (col < 2)       m = 1.f - (float)bc_disp[(size_t)node * 2 + col];
            else if (col == 2) m = 1.f - (float)bc_rot[node];
            else               m = (float)face_mask[(size_t)node * 4 + (col - 3) / 3];
            pred[(size_t)node * 15 + col] = (acc[rt][r] + bv) * m;
          }
        }
    }
  }
}

// ---------------------------------------------------------------------------
extern "C" void kernel_launch(void* const* d_in, const int* in_sizes, int n_in,
                              void* d_out, int out_size, void* d_ws, size_t ws_size,
                              hipStream_t stream) {
  const float* x         = (const float*)d_in[0];
  const float* coords    = (const float*)d_in[1];
  const float* edge_attr = (const float*)d_in[2];
  const int*   eidx      = (const int*)d_in[3];
  const int*   bc_disp   = (const int*)d_in[4];
  const int*   bc_rot    = (const int*)d_in[5];
  const int*   face_mask = (const int*)d_in[6];

  const int N = in_sizes[0] / 9;
  const int E = in_sizes[2] / 10;
  const int* src = eidx;
  const int* dst = eidx + E;

  // JAX pytree flatten order (dict keys sorted): dec, edge_enc, mp, node_enc
  const float* decW0 = (const float*)d_in[7];
  const float* decB0 = (const float*)d_in[8];
  const float* decW1 = (const float*)d_in[9];
  const float* decB1 = (const float*)d_in[10];
  const float* decW2 = (const float*)d_in[11];
  const float* decB2 = (const float*)d_in[12];
  const float* eeW0  = (const float*)d_in[13];
  const float* eeB0  = (const float*)d_in[14];
  const float* eeW1  = (const float*)d_in[15];
  const float* eeB1  = (const float*)d_in[16];
  // mp layer l leaves at 17 + 8*l: eW0,eB0,eW1,eB1,nW0,nB0,nW1,nB1
  const float* neW0  = (const float*)d_in[65];
  const float* neB0  = (const float*)d_in[66];
  const float* neW1  = (const float*)d_in[67];
  const float* neB1  = (const float*)d_in[68];

  // --- workspace carve-up: packed bf16 weights, then f32 h / e / agg ---
  __bf16* wbase = (__bf16*)d_ws;
  size_t off = 0;
  auto carve = [&](size_t elems) { __bf16* p = wbase + off; off += elems; return p; };
  __bf16* p_neW0 = carve(8 * 1 * 512);
  __bf16* p_neW1 = carve(8 * 4 * 512);
  __bf16* p_eeW0 = carve(8 * 1 * 512);
  __bf16* p_eeW1 = carve(8 * 4 * 512);
  __bf16 *p_eW0[6], *p_eW1[6], *p_nW0[6], *p_nW1[6];
  for (int l = 0; l < 6; ++l) {
    p_eW0[l] = carve(8 * 12 * 512);
    p_eW1[l] = carve(8 * 4 * 512);
    p_nW0[l] = carve(8 * 8 * 512);
    p_nW1[l] = carve(8 * 4 * 512);
  }
  __bf16* p_dW0 = carve(8 * 4 * 512);
  __bf16* p_dW1 = carve(4 * 4 * 512);
  __bf16* p_dW2 = carve(1 * 2 * 512);
  size_t woff_bytes = (off * sizeof(__bf16) + 255) & ~(size_t)255;
  float* hbuf   = (float*)((char*)d_ws + woff_bytes);
  float* ebuf   = hbuf + (size_t)N * 128;
  float* aggbuf = ebuf + (size_t)E * 128;

  auto pack = [&](const float* W, __bf16* outp, int Kin, int Nout, int KC, int NT) {
    int total = NT * KC * 512;
    pack_w<<<(total + 255) / 256, 256, 0, stream>>>(W, outp, Kin, Nout, KC, NT);
  };
  pack(neW0, p_neW0, 9, 128, 1, 8);
  pack(neW1, p_neW1, 128, 128, 4, 8);
  pack(eeW0, p_eeW0, 10, 128, 1, 8);
  pack(eeW1, p_eeW1, 128, 128, 4, 8);
  for (int l = 0; l < 6; ++l) {
    pack((const float*)d_in[17 + 8 * l + 0], p_eW0[l], 384, 128, 12, 8);
    pack((const float*)d_in[17 + 8 * l + 2], p_eW1[l], 128, 128, 4, 8);
    pack((const float*)d_in[17 + 8 * l + 4], p_nW0[l], 256, 128, 8, 8);
    pack((const float*)d_in[17 + 8 * l + 6], p_nW1[l], 128, 128, 4, 8);
  }
  pack(decW0, p_dW0, 128, 128, 4, 8);
  pack(decW1, p_dW1, 128, 64, 4, 4);
  pack(decW2, p_dW2, 64, 15, 2, 1);

  const int nb64 = (N + 63) / 64;
  enc_nodes<<<nb64, 256, 0, stream>>>(x, coords, p_neW0, neB0, p_neW1, neB1, hbuf, N);
  enc_edges<<<(E + 63) / 64, 256, 0, stream>>>(edge_attr, p_eeW0, eeB0, p_eeW1, eeB1, ebuf, E);

  for (int l = 0; l < 6; ++l) {
    const float* eB0 = (const float*)d_in[17 + 8 * l + 1];
    const float* eB1 = (const float*)d_in[17 + 8 * l + 3];
    const float* nB0 = (const float*)d_in[17 + 8 * l + 5];
    const float* nB1 = (const float*)d_in[17 + 8 * l + 7];
    int n4 = (N * 128) / 4;
    zero_f32v4<<<(n4 + 255) / 256, 256, 0, stream>>>((float4*)aggbuf, n4);
    edge_mp<<<(E + 31) / 32, 256, 0, stream>>>(hbuf, ebuf, src, dst,
                                               p_eW0[l], eB0, p_eW1[l], eB1, aggbuf, E);
    node_mp<<<nb64, 256, 0, stream>>>(hbuf, aggbuf, p_nW0[l], nB0, p_nW1[l], nB1, N);
  }
  decode<<<nb64, 256, 0, stream>>>(hbuf, p_dW0, decB0, p_dW1, decB1, p_dW2, decB2,
                                   bc_disp, bc_rot, face_mask, (float*)d_out, N);
}
